// TransformerBlock_85856396247666
// MI455X (gfx1250) — compile-verified, matrix operand order checked by reading, not run on hardware
//
#include <hip/hip_runtime.h>

// ---------------------------------------------------------------------------
// Types for CDNA5 WMMA (wave32): V_WMMA_F32_16X16X32_BF16
// ---------------------------------------------------------------------------
typedef __bf16 bf16_t;
typedef __attribute__((ext_vector_type(16))) __bf16 v16bf;
typedef __attribute__((ext_vector_type(8)))  __bf16 v8bf;
typedef __attribute__((ext_vector_type(8)))  float  v8f;
typedef __attribute__((ext_vector_type(4)))  unsigned int v4u;
typedef __attribute__((ext_vector_type(8)))  int v8i;
typedef __attribute__((ext_vector_type(4)))  int v4i;

#define B_  4
#define S_  2048
#define E_  1024
#define H_  16
#define D_  64
#define F_  4096
#define M_  (B_*S_)

#define MODE_F32        0
#define MODE_BF16       1
#define MODE_RELU_BF16  2
#define MODE_QKV        3

__device__ __forceinline__ v8f wmma_bf16(v16bf a, v16bf b, v8f c) {
  return __builtin_amdgcn_wmma_f32_16x16x32_bf16(
      /*neg_a=*/false, a, /*neg_b=*/false, b,
      /*c_mod=*/(short)0, c, /*reuse_a=*/false, /*reuse_b=*/false);
}

// ---------------------------------------------------------------------------
// gfx1250 async global->LDS copies (ASYNCcnt) and waits
// ---------------------------------------------------------------------------
#define WAIT_ASYNC(n) asm volatile("s_wait_asynccnt " #n ::: "memory")

// copy 64B per lane: 4 x GLOBAL_LOAD_ASYNC_TO_LDS_B128 (same INST_OFFSET is
// applied to both the LDS and the global address per the ISA)
__device__ __forceinline__ void async_cp64(unsigned int lds, const bf16_t* g) {
  asm volatile(
      "global_load_async_to_lds_b128 %0, %1, off\n\t"
      "global_load_async_to_lds_b128 %0, %1, off offset:16\n\t"
      "global_load_async_to_lds_b128 %0, %1, off offset:32\n\t"
      "global_load_async_to_lds_b128 %0, %1, off offset:48"
      :: "v"(lds), "v"(g) : "memory");
}
// copy 32B per lane
__device__ __forceinline__ void async_cp32(unsigned int lds, const bf16_t* g) {
  asm volatile(
      "global_load_async_to_lds_b128 %0, %1, off\n\t"
      "global_load_async_to_lds_b128 %0, %1, off offset:16"
      :: "v"(lds), "v"(g) : "memory");
}

__device__ __forceinline__ unsigned int lds_addr_of(const void* p) {
  return (unsigned int)(unsigned long long)p;  // generic LDS addr: low 32 bits
}

// ---------------------------------------------------------------------------
// TDM: one 2D-tile DMA (Tensor Data Mover) load into LDS. Descriptor per
// CDNA5 ISA  (D# group0 128b + group1 256b, groups2/3 = 0)
// data_size = 2 bytes, tile rows land contiguously (row stride = tile_d0).
// This toolchain exposes the 6-arg builtin (extra int32x8 before cpol).
// ---------------------------------------------------------------------------
__device__ __forceinline__ void tdm_load_2d(unsigned int lds_addr, const void* gaddr,
                                            unsigned int tensor_d0, unsigned int tensor_d1,
                                            unsigned int tile_d0, unsigned int tile_d1,
                                            unsigned int stride_d0)
{
  const unsigned long long ga = (unsigned long long)gaddr;
  v4u g0;
  g0[0] = 1u;                                            // count=1 (user D#)
  g0[1] = lds_addr;                                      // lds_addr[31:0]
  g0[2] = (unsigned int)ga;                              // global_addr[31:0]
  g0[3] = (unsigned int)((ga >> 32) & 0x1FFFFFFull) | (2u << 30); // [56:32] | type=2
  v8i g1;
  g1[0] = (int)(1u << 16);                               // data_size=1 (2B), wg_mask=0
  g1[1] = (int)((tensor_d0 & 0xFFFFu) << 16);            // tensor_dim0[15:0]
  g1[2] = (int)(((tensor_d0 >> 16) & 0xFFFFu) | ((tensor_d1 & 0xFFFFu) << 16));
  g1[3] = (int)(((tensor_d1 >> 16) & 0xFFFFu) | ((tile_d0 & 0xFFFFu) << 16));
  g1[4] = (int)(tile_d1 & 0xFFFFu);                      // tile_dim1 (tile_dim2=0)
  g1[5] = (int)stride_d0;                                // tensor_dim0_stride[31:0]
  g1[6] = 0;
  g1[7] = 0;
  const v4i z4 = (v4i){0, 0, 0, 0};
  const v8i z8 = (v8i){0, 0, 0, 0, 0, 0, 0, 0};
  __builtin_amdgcn_tensor_load_to_lds(g0, g1, z4, z4, z8, 0);
}

// Load a 16x32 bf16 A/B-operand fragment from an M-/N-major tile with row
// stride `ld` (elements). 16-bit A layout: lanes 0-15 hold M=lane,
// K = 0..7 & 16..23 ; lanes 16-31 hold M=lane-16, K = 8..15 & 24..31.
__device__ __forceinline__ v16bf frag_ld(const bf16_t* p, int ld) {
  const int lane = threadIdx.x & 31;
  const int lr   = lane & 15;
  const int c    = (lane >> 4) << 3;
  const bf16_t* q = p + lr * ld + c;
  union { v16bf v; v8bf h[2]; } u;
  u.h[0] = *(const v8bf*)(q);
  u.h[1] = *(const v8bf*)(q + 16);
  return u.v;
}

// half-wave (16 lane) reductions: C-tile rows live across lanes sharing lane>>4
__device__ __forceinline__ float hred_max(float v) {
#pragma unroll
  for (int m = 1; m < 16; m <<= 1) v = fmaxf(v, __shfl_xor(v, m, 32));
  return v;
}
__device__ __forceinline__ float hred_sum(float v) {
#pragma unroll
  for (int m = 1; m < 16; m <<= 1) v += __shfl_xor(v, m, 32);
  return v;
}

// ---------------------------------------------------------------------------
// Generic GEMM:  C[M][N] = A[M][K](bf16) * Wt[N][K](bf16, N-major) + bias
// 128x128 block tile, BK=64, 8 waves, wave computes 64x32 of C.
// A tile staged via async-to-LDS (all waves), B tile via TDM (wave 0),
// double buffered so the next tile streams in during WMMA compute.
// ---------------------------------------------------------------------------
#define BM 128
#define BN 128
#define BK 64

__global__ __launch_bounds__(256) void gemm_bf16_kernel(
    const bf16_t* __restrict__ A, const bf16_t* __restrict__ Bt,
    const float* __restrict__ bias, void* __restrict__ out,
    int M, int N, int K, int mode)
{
  __shared__ bf16_t As[2][BM * BK];
  __shared__ bf16_t Bs[2][BM * BK];

  const int tid  = threadIdx.x;
  const int wave = tid >> 5;
  const int lane = tid & 31;
  const int lr   = lane & 15;
  const int half = lane >> 4;
  const int wm = (wave & 1) * 64;   // wave row offset
  const int wn = (wave >> 1) * 32;  // wave col offset
  const int m0 = blockIdx.y * BM;
  const int n0 = blockIdx.x * BN;

  const int srow = tid >> 1;        // 0..127
  const int scol = (tid & 1) * 32;  // 0 or 32 (32 bf16 = 64B per thread)
  const bf16_t* gA = A + (size_t)(m0 + srow) * K + scol;
  const unsigned int ldsA0 = lds_addr_of(&As[0][srow * BK + scol]);
  const unsigned int ldsA1 = lds_addr_of(&As[1][srow * BK + scol]);
  const unsigned int ldsB0 = lds_addr_of(&Bs[0][0]);
  const unsigned int ldsB1 = lds_addr_of(&Bs[1][0]);

  v8f acc[4][2];
#pragma unroll
  for (int i = 0; i < 4; ++i)
#pragma unroll
    for (int j = 0; j < 2; ++j) acc[i][j] = (v8f){0,0,0,0,0,0,0,0};

  // prologue: stage tile 0
  async_cp64(ldsA0, gA);
  if (wave == 0)
    tdm_load_2d(ldsB0, Bt + (size_t)n0 * K, K, N, BK, BN, K);

  const int T = K / BK;
  int cur = 0;
  for (int t = 0; t < T; ++t) {
    const bool more = (t + 1) < T;
    if (more) {  // kick next tile before waiting on current
      const int k1 = (t + 1) * BK;
      async_cp64(cur ? ldsA0 : ldsA1, gA + k1);
      if (wave == 0)
        tdm_load_2d(cur ? ldsB0 : ldsB1, Bt + (size_t)n0 * K + k1, K, N, BK, BN, K);
    }
    if (more) { WAIT_ASYNC(4); } else { WAIT_ASYNC(0); }  // current A done (in-order)
    if (wave == 0) {
      if (more) __builtin_amdgcn_s_wait_tensorcnt(1);     // current B done
      else      __builtin_amdgcn_s_wait_tensorcnt(0);
    }
    __syncthreads();

    const bf16_t* as = &As[cur][0];
    const bf16_t* bs = &Bs[cur][0];
#pragma unroll
    for (int kk = 0; kk < BK; kk += 32) {
      v16bf af[4], bfv[2];
#pragma unroll
      for (int i = 0; i < 4; ++i) af[i]  = frag_ld(as + (wm + i*16) * BK + kk, BK);
#pragma unroll
      for (int j = 0; j < 2; ++j) bfv[j] = frag_ld(bs + (wn + j*16) * BK + kk, BK);
#pragma unroll
      for (int i = 0; i < 4; ++i)
#pragma unroll
        for (int j = 0; j < 2; ++j) acc[i][j] = wmma_bf16(af[i], bfv[j], acc[i][j]);
    }
    __syncthreads();   // compute done: safe for next iteration's DMA into this buf
    cur ^= 1;
  }

  // Epilogue: C layout => lane lr = col, VGPR r = row (r + 8*half) within tile
#pragma unroll
  for (int i = 0; i < 4; ++i) {
#pragma unroll
    for (int j = 0; j < 2; ++j) {
      const int ncol = n0 + wn + j*16 + lr;
      const float bv = bias ? bias[ncol] : 0.f;
#pragma unroll
      for (int r = 0; r < 8; ++r) {
        const int row = m0 + wm + i*16 + half*8 + r;
        float v = acc[i][j][r] + bv;
        if (mode == MODE_F32) {
          ((float*)out)[(size_t)row * N + ncol] = v;
        } else if (mode == MODE_BF16) {
          ((bf16_t*)out)[(size_t)row * N + ncol] = (bf16_t)v;
        } else if (mode == MODE_RELU_BF16) {
          v = fmaxf(v, 0.f);
          ((bf16_t*)out)[(size_t)row * N + ncol] = (bf16_t)v;
        } else { // MODE_QKV: [B*S, E] -> [B,H,S,D]
          const int b = row >> 11, s = row & (S_ - 1);
          const int h = ncol >> 6, d = ncol & (D_ - 1);
          ((bf16_t*)out)[(((size_t)(b*H_ + h))*S_ + s)*D_ + d] = (bf16_t)v;
        }
      }
    }
  }
}

// ---------------------------------------------------------------------------
// Flash attention: block = (b,h) x 128 q rows; wave owns a 16-row band.
// K tile staged via async-to-LDS; V staged transposed via VALU path.
// ---------------------------------------------------------------------------
__global__ __launch_bounds__(256) void attn_kernel(
    const bf16_t* __restrict__ Q, const bf16_t* __restrict__ Kmat,
    const bf16_t* __restrict__ V, bf16_t* __restrict__ Out)
{
  __shared__ bf16_t Ks [64 * 64];     // [k][d]
  __shared__ bf16_t Vts[64 * 64];     // [d][k] (transposed V)
  __shared__ bf16_t Pw [8][16 * 64];  // per-wave P tile for C->A relayout

  const int tid  = threadIdx.x;
  const int wave = tid >> 5;
  const int lane = tid & 31;
  const int lr   = lane & 15;
  const int half = lane >> 4;
  const int bh = blockIdx.y;
  const int q0 = blockIdx.x * 128;
  const size_t base = (size_t)bh * S_ * D_;
  const bf16_t* qp = Q    + base;
  const bf16_t* kp = Kmat + base;
  const bf16_t* vp = V    + base;
  const int mband = wave * 16;

  // Q fragments straight from global (A layout is contiguous per row)
  const v16bf aq0 = frag_ld(qp + (size_t)(q0 + mband) * D_ + 0,  D_);
  const v16bf aq1 = frag_ld(qp + (size_t)(q0 + mband) * D_ + 32, D_);

  float mrow[8], lrow[8];
  v8f o[4];
#pragma unroll
  for (int r = 0; r < 8; ++r) { mrow[r] = -3.0e38f; lrow[r] = 0.f; }
#pragma unroll
  for (int dt = 0; dt < 4; ++dt) o[dt] = (v8f){0,0,0,0,0,0,0,0};

  const int srow = tid >> 2;        // 0..63
  const int scol = (tid & 3) * 16;  // 16 elems = 32B per thread
  const unsigned int ldsK = lds_addr_of(&Ks[srow * 64 + scol]);

  for (int k0 = 0; k0 < S_; k0 += 64) {
    __syncthreads();
    // async K tile copy (overlaps with the V transpose below)
    async_cp32(ldsK, kp + (size_t)(k0 + srow) * D_ + scol);
    { // V tile staged transposed
      const uint4* gv = (const uint4*)(vp + (size_t)(k0 + srow) * D_ + scol);
      alignas(16) bf16_t tmp[16];
      *(uint4*)(tmp)     = gv[0];
      *(uint4*)(tmp + 8) = gv[1];
#pragma unroll
      for (int i = 0; i < 16; ++i) Vts[(scol + i) * 64 + srow] = tmp[i];
    }
    WAIT_ASYNC(0);
    __syncthreads();

    // S = Q * K^T  (B operand = rows of K, already N-major in Ks)
    v8f sc[4];
#pragma unroll
    for (int nt = 0; nt < 4; ++nt) {
      sc[nt] = (v8f){0,0,0,0,0,0,0,0};
      v16bf b0 = frag_ld(Ks + (nt*16) * 64 + 0,  64);
      v16bf b1 = frag_ld(Ks + (nt*16) * 64 + 32, 64);
      sc[nt] = wmma_bf16(aq0, b0, sc[nt]);
      sc[nt] = wmma_bf16(aq1, b1, sc[nt]);
    }
#pragma unroll
    for (int nt = 0; nt < 4; ++nt)
#pragma unroll
      for (int r = 0; r < 8; ++r) sc[nt][r] *= 0.125f;  // 1/sqrt(64)

    // online softmax (row stats reduce within the 16-lane half)
    float nm[8], alpha[8];
#pragma unroll
    for (int r = 0; r < 8; ++r) {
      float mx = fmaxf(fmaxf(sc[0][r], sc[1][r]), fmaxf(sc[2][r], sc[3][r]));
      mx = hred_max(mx);
      nm[r]    = fmaxf(mrow[r], mx);
      alpha[r] = __expf(mrow[r] - nm[r]);
      mrow[r]  = nm[r];
    }
#pragma unroll
    for (int nt = 0; nt < 4; ++nt)
#pragma unroll
      for (int r = 0; r < 8; ++r) sc[nt][r] = __expf(sc[nt][r] - nm[r]);
#pragma unroll
    for (int r = 0; r < 8; ++r) {
      float s = sc[0][r] + sc[1][r] + sc[2][r] + sc[3][r];
      s = hred_sum(s);
      lrow[r] = lrow[r] * alpha[r] + s;
    }
#pragma unroll
    for (int dt = 0; dt < 4; ++dt)
#pragma unroll
      for (int r = 0; r < 8; ++r) o[dt][r] *= alpha[r];

    // P: C layout -> LDS -> A layout (wave-private, LDS in-order per wave)
    bf16_t* pw = &Pw[wave][0];
#pragma unroll
    for (int nt = 0; nt < 4; ++nt)
#pragma unroll
      for (int r = 0; r < 8; ++r)
        pw[(half*8 + r) * 64 + nt*16 + lr] = (bf16_t)sc[nt][r];
    asm volatile("" ::: "memory");  // keep compiler from hoisting the reloads

    v16bf pa0 = frag_ld(pw + 0,  64);
    v16bf pa1 = frag_ld(pw + 32, 64);
#pragma unroll
    for (int dt = 0; dt < 4; ++dt) {
      v16bf vb0 = frag_ld(Vts + (dt*16) * 64 + 0,  64);
      v16bf vb1 = frag_ld(Vts + (dt*16) * 64 + 32, 64);
      o[dt] = wmma_bf16(pa0, vb0, o[dt]);
      o[dt] = wmma_bf16(pa1, vb1, o[dt]);
    }
  }

  // write attended back to [B,S,E] (bf16, feeds O-projection GEMM)
  const int b = bh >> 4, h = bh & 15;
#pragma unroll
  for (int dt = 0; dt < 4; ++dt)
#pragma unroll
    for (int r = 0; r < 8; ++r) {
      const int s = q0 + mband + half*8 + r;
      const float val = o[dt][r] / lrow[r];
      Out[((size_t)(b * S_ + s)) * E_ + h*D_ + dt*16 + lr] = (bf16_t)val;
    }
}

// ---------------------------------------------------------------------------
// Fused residual + LayerNorm (f32 math); optional f32 and bf16 outputs.
// ---------------------------------------------------------------------------
__global__ __launch_bounds__(256) void add_ln_kernel(
    const float* __restrict__ a, const float* __restrict__ bres,
    const float* __restrict__ g, const float* __restrict__ be,
    float* __restrict__ outf, bf16_t* __restrict__ outb)
{
  __shared__ float rs[256], rq[256];
  const int row = blockIdx.x;
  const int tid = threadIdx.x;
  const size_t off = (size_t)row * E_;
  float x[4], s = 0.f, s2 = 0.f;
#pragma unroll
  for (int i = 0; i < 4; ++i) {
    const int c = tid + i*256;
    const float v = a[off + c] + bres[off + c];
    x[i] = v; s += v; s2 += v*v;
  }
  rs[tid] = s; rq[tid] = s2;
  __syncthreads();
  for (int st = 128; st > 0; st >>= 1) {
    if (tid < st) { rs[tid] += rs[tid+st]; rq[tid] += rq[tid+st]; }
    __syncthreads();
  }
  const float mu   = rs[0] * (1.f/(float)E_);
  const float var  = rq[0] * (1.f/(float)E_) - mu*mu;
  const float rstd = rsqrtf(var + 1e-5f);
#pragma unroll
  for (int i = 0; i < 4; ++i) {
    const int c = tid + i*256;
    const float y = (x[i] - mu) * rstd * g[c] + be[c];
    if (outf) outf[off + c] = y;
    if (outb) outb[off + c] = (bf16_t)y;
  }
}

// ---------------------------------------------------------------------------
// Conversions
// ---------------------------------------------------------------------------
__global__ __launch_bounds__(256) void cvt_bf16_kernel(
    const float* __restrict__ s, bf16_t* __restrict__ d, int n)
{
  const int i = (blockIdx.x * 256 + threadIdx.x) * 4;
  if (i + 3 < n) {
    const float4 v = *(const float4*)(s + i);
    d[i+0] = (bf16_t)v.x; d[i+1] = (bf16_t)v.y;
    d[i+2] = (bf16_t)v.z; d[i+3] = (bf16_t)v.w;
  }
}

// Wt[n][k] = (bf16) W[k][n]   (W is K x N row-major)
__global__ __launch_bounds__(256) void tcvt_kernel(
    const float* __restrict__ W, bf16_t* __restrict__ Wt, int K, int N)
{
  __shared__ float t[32][33];
  const int bn = blockIdx.x * 32, bk = blockIdx.y * 32;
  const int tx = threadIdx.x, ty = threadIdx.y;
#pragma unroll
  for (int i = 0; i < 32; i += 8)
    t[ty + i][tx] = W[(size_t)(bk + ty + i) * N + bn + tx];
  __syncthreads();
#pragma unroll
  for (int i = 0; i < 32; i += 8)
    Wt[(size_t)(bn + ty + i) * K + bk + tx] = (bf16_t)t[tx][ty + i];
}

// ---------------------------------------------------------------------------
// Host-side launch sequence
// ---------------------------------------------------------------------------
extern "C" void kernel_launch(void* const* d_in, const int* in_sizes, int n_in,
                              void* d_out, int out_size, void* d_ws, size_t ws_size,
                              hipStream_t stream)
{
  (void)in_sizes; (void)n_in; (void)out_size; (void)ws_size;
  const float* x    = (const float*)d_in[0];
  const float* qw   = (const float*)d_in[1];
  const float* qb   = (const float*)d_in[2];
  const float* kw   = (const float*)d_in[3];
  const float* kb   = (const float*)d_in[4];
  const float* vw   = (const float*)d_in[5];
  const float* vb   = (const float*)d_in[6];
  const float* ow   = (const float*)d_in[7];
  const float* ob   = (const float*)d_in[8];
  const float* ln1g = (const float*)d_in[9];
  const float* ln1b = (const float*)d_in[10];
  const float* w1   = (const float*)d_in[11];
  const float* b1   = (const float*)d_in[12];
  const float* w2   = (const float*)d_in[13];
  const float* b2   = (const float*)d_in[14];
  const float* ln2g = (const float*)d_in[15];
  const float* ln2b = (const float*)d_in[16];

  char* ws = (char*)d_ws;
  const size_t MB = 1ull << 20;
  bf16_t* xb     = (bf16_t*)(ws + 0);        // 16 MB
  bf16_t* qwt    = (bf16_t*)(ws + 16*MB);    //  2 MB
  bf16_t* kwt    = (bf16_t*)(ws + 18*MB);    //  2 MB
  bf16_t* vwt    = (bf16_t*)(ws + 20*MB);    //  2 MB
  bf16_t* owt    = (bf16_t*)(ws + 22*MB);    //  2 MB
  bf16_t* w1t    = (bf16_t*)(ws + 24*MB);    //  8 MB
  bf16_t* w2t    = (bf16_t*)(ws + 32*MB);    //  8 MB
  bf16_t* qbuf   = (bf16_t*)(ws + 40*MB);    // 16 MB  [B,H,S,D]
  bf16_t* kbuf   = (bf16_t*)(ws + 56*MB);    // 16 MB
  bf16_t* vbuf   = (bf16_t*)(ws + 72*MB);    // 16 MB
  bf16_t* attn_b = (bf16_t*)(ws + 88*MB);    // 16 MB  attended bf16
  float*  attn_f = (float*) (ws + 104*MB);   // 32 MB  attn proj out f32
  float*  hF     = (float*) (ws + 136*MB);   // 32 MB  h (f32 residual)
  bf16_t* hB     = (bf16_t*)(ws + 168*MB);   // 16 MB  h (bf16 GEMM input)
  bf16_t* ff1    = (bf16_t*)(ws + 184*MB);   // 64 MB  relu(h@w1+b1) bf16
  float*  ff2    = (float*) (ws + 104*MB);   // alias attn_f (free by then)

  const dim3 blk(256);
  const dim3 tblk(32, 8);

  cvt_bf16_kernel<<<(M_*E_)/1024, blk, 0, stream>>>(x, xb, M_*E_);
  tcvt_kernel<<<dim3(E_/32, E_/32), tblk, 0, stream>>>(qw, qwt, E_, E_);
  tcvt_kernel<<<dim3(E_/32, E_/32), tblk, 0, stream>>>(kw, kwt, E_, E_);
  tcvt_kernel<<<dim3(E_/32, E_/32), tblk, 0, stream>>>(vw, vwt, E_, E_);
  tcvt_kernel<<<dim3(E_/32, E_/32), tblk, 0, stream>>>(ow, owt, E_, E_);
  tcvt_kernel<<<dim3(F_/32, E_/32), tblk, 0, stream>>>(w1, w1t, E_, F_);
  tcvt_kernel<<<dim3(E_/32, F_/32), tblk, 0, stream>>>(w2, w2t, F_, E_);

  // Q/K/V projections, scattered straight into [B,H,S,D] bf16
  gemm_bf16_kernel<<<dim3(E_/128, M_/128), blk, 0, stream>>>(xb, qwt, qb, qbuf, M_, E_, E_, MODE_QKV);
  gemm_bf16_kernel<<<dim3(E_/128, M_/128), blk, 0, stream>>>(xb, kwt, kb, kbuf, M_, E_, E_, MODE_QKV);
  gemm_bf16_kernel<<<dim3(E_/128, M_/128), blk, 0, stream>>>(xb, vwt, vb, vbuf, M_, E_, E_, MODE_QKV);

  attn_kernel<<<dim3(S_/128, B_*H_), blk, 0, stream>>>(qbuf, kbuf, vbuf, attn_b);

  gemm_bf16_kernel<<<dim3(E_/128, M_/128), blk, 0, stream>>>(attn_b, owt, ob, attn_f, M_, E_, E_, MODE_F32);
  add_ln_kernel<<<M_, blk, 0, stream>>>(x, attn_f, ln1g, ln1b, hF, hB);

  gemm_bf16_kernel<<<dim3(F_/128, M_/128), blk, 0, stream>>>(hB,  w1t, b1, ff1, M_, F_, E_, MODE_RELU_BF16);
  gemm_bf16_kernel<<<dim3(E_/128, M_/128), blk, 0, stream>>>(ff1, w2t, b2, ff2, M_, E_, F_, MODE_F32);
  add_ln_kernel<<<M_, blk, 0, stream>>>(hF, ff2, ln2g, ln2b, (float*)d_out, nullptr);
}
